// SpatialRelationExtractor_48808008351770
// MI455X (gfx1250) — matrix operand order repaired and checked by reading, not compiled, not run on hardware
//
#include <hip/hip_runtime.h>

typedef __attribute__((ext_vector_type(16))) _Float16 v16h;
typedef __attribute__((ext_vector_type(8)))  _Float16 v8h;
typedef __attribute__((ext_vector_type(8)))  float    v8f;
typedef __attribute__((ext_vector_type(4)))  float    v4f;

#define BQ 8
#define NQ 128
#define DQ 512       // feature dim / K of GEMM1
#define C1 256       // hidden of GEMM2 / cols of GEMM1
#define TN 8         // n per workgroup
#define TM 16        // m per workgroup
#define ROWS 128     // TN*TM pair-rows per workgroup
#define H2S 264      // padded h2 row stride (f16): 528B, 16B-aligned, bank-skewed
#define TS  516      // padded A/Bv tile row stride (f32): 2064B, 16B-aligned, bank-skewed

// ---------------------------------------------------------------------------
// Kernel 1: A[b,n,k] = feat@W1a + pos@W1c + b1 ; Bv[b,m,k] = feat@W1b - pos@W1c
// ---------------------------------------------------------------------------
__global__ __launch_bounds__(256) void prep_rows(
    const float* __restrict__ feat, const float* __restrict__ pos,
    const float* __restrict__ W1, const float* __restrict__ b1,
    float* __restrict__ Arows, float* __restrict__ Brows)
{
  __shared__ float sf[DQ];
  const int row = blockIdx.x;            // b*NQ + n
  const int tid = threadIdx.x;
  sf[tid]       = feat[(size_t)row * DQ + tid];
  sf[tid + 256] = feat[(size_t)row * DQ + 256 + tid];
  __syncthreads();
  const float p0 = pos[row * 2 + 0], p1 = pos[row * 2 + 1];
  #pragma unroll
  for (int kk = 0; kk < 2; ++kk) {
    const int k = tid + kk * 256;
    float a = 0.f, bb = 0.f;
    #pragma unroll 8
    for (int d = 0; d < DQ; ++d) {
      const float f = sf[d];
      a  += f * W1[(size_t)d * DQ + k];
      bb += f * W1[(size_t)(DQ + d) * DQ + k];
    }
    const float pc = p0 * W1[(size_t)(2 * DQ) * DQ + k] +
                     p1 * W1[(size_t)(2 * DQ + 1) * DQ + k];
    Arows[(size_t)row * DQ + k] = a + pc + b1[k];
    Brows[(size_t)row * DQ + k] = bb - pc;
  }
}

// ---------------------------------------------------------------------------
// Kernel 2/3: repack weights to f16, (N,K) layout for direct B-fragment loads
// ---------------------------------------------------------------------------
__global__ __launch_bounds__(256) void prep_w2t(const float* __restrict__ W2,
                                                _Float16* __restrict__ W2t)
{
  const int t = blockIdx.x * 256 + threadIdx.x;   // t < C1*DQ
  const int n = t >> 9, k = t & 511;              // W2t[n][k] = W2[k][n]
  W2t[t] = (_Float16)W2[(size_t)k * C1 + n];
}

__global__ __launch_bounds__(256) void prep_w3t(const float* __restrict__ W3,
                                                _Float16* __restrict__ W3t)
{
  const int t = blockIdx.x * 256 + threadIdx.x;   // t < 16*C1
  const int n = t >> 8, k = t & 255;              // zero-pad cols 4..15
  W3t[t] = (n < 4) ? (_Float16)W3[(size_t)k * 4 + n] : (_Float16)0.f;
}

// ---------------------------------------------------------------------------
// Kernel 4: fused pairwise MLP.  8 waves; wave w: col-half (w&1),
// n_local tiles {2*(w>>1), 2*(w>>1)+1}.  A/Bv tiles staged in LDS; W2
// B-fragments streamed from L2; GEMM2 via LDS-staged h2 (unioned LDS).
// ---------------------------------------------------------------------------
__global__ __launch_bounds__(256) void fused(
    const float* __restrict__ Arows, const float* __restrict__ Brows,
    const _Float16* __restrict__ W2t, const _Float16* __restrict__ W3t,
    const float* __restrict__ b2, const float* __restrict__ b3,
    float* __restrict__ out)
{
  extern __shared__ char lds[];
  float*    sB  = (float*)lds;                         // 16 x TS f32 (33,024 B)
  float*    sA  = (float*)(lds + 16 * TS * 4);         //  8 x TS f32 (16,512 B)
  _Float16* h2s = (_Float16*)lds;                      // ROWS x H2S f16 (union)

  const int tid  = threadIdx.x;
  const int w    = tid >> 5;
  const int lane = tid & 31;
  const int nl   = lane & 15;          // fragment row/col within 16
  const int lh   = lane >> 4;          // lane half -> K offset select
  const int koff = lh * 8;

  const int blk = blockIdx.x;
  const int b   = blk >> 7;            // 128 tiles per batch (16 n-tiles * 8 m-tiles)
  const int r   = blk & 127;
  const int n0  = (r >> 3) * TN;
  const int m0  = (r & 7) * TM;

  const int chalf = w & 1;             // which 128 of the 256 cols
  const int rtb   = (w >> 1) * 2;      // base n_local (row-tile)

  // Warm L2 for this lane's W2t column rows (CDNA5 global_prefetch path).
  #pragma unroll
  for (int ct = 0; ct < 8; ++ct) {
    const _Float16* p = W2t + (size_t)(chalf * 128 + ct * 16 + nl) * DQ;
    __builtin_prefetch(p, 0, 3);
    __builtin_prefetch(p + 256, 0, 3);
  }

  // per-lane b2 for this lane's 8 column tiles
  float b2r[8];
  #pragma unroll
  for (int ct = 0; ct < 8; ++ct) b2r[ct] = b2[chalf * 128 + ct * 16 + nl];

  // ---- cooperative stage of A (8x512) and Bv (16x512) tiles into LDS ----
  {
    const float* gB = Brows + (size_t)(b * NQ + m0) * DQ;
    const float* gA = Arows + (size_t)(b * NQ + n0) * DQ;
    #pragma unroll
    for (int i = tid * 4; i < 16 * DQ; i += 256 * 4) {
      const int rr = i >> 9, k = i & 511;
      *(v4f*)(sB + rr * TS + k) = *(const v4f*)(gB + rr * DQ + k);
    }
    #pragma unroll
    for (int i = tid * 4; i < 8 * DQ; i += 256 * 4) {
      const int rr = i >> 9, k = i & 511;
      *(v4f*)(sA + rr * TS + k) = *(const v4f*)(gA + rr * DQ + k);
    }
  }
  __syncthreads();

  const float* ArowL0 = sA + (rtb)     * TS;   // wave-uniform rows
  const float* ArowL1 = sA + (rtb + 1) * TS;
  const float* BrowL  = sB + nl * TS;          // per-lane row m

  v8f acc[2][8];
  #pragma unroll
  for (int i = 0; i < 2; ++i)
    #pragma unroll
    for (int j = 0; j < 8; ++j)
      #pragma unroll
      for (int e = 0; e < 8; ++e) acc[i][j][e] = 0.f;

  // ---- GEMM1: h(128x512) @ W2(512x256), h built on the fly from LDS ----
  #pragma unroll 2
  for (int ks = 0; ks < DQ / 32; ++ks) {
    const int kb = ks * 32;
    const int k0 = kb + koff;         // dwords 0..3 of A frag
    const int k1 = kb + 16 + koff;    // dwords 4..7 of A frag

    const v4f ba0 = *(const v4f*)(BrowL + k0);
    const v4f ba1 = *(const v4f*)(BrowL + k0 + 4);
    const v4f bb0 = *(const v4f*)(BrowL + k1);
    const v4f bb1 = *(const v4f*)(BrowL + k1 + 4);

    v16h a0, a1;
    {
      const v4f x0 = *(const v4f*)(ArowL0 + k0);
      const v4f x1 = *(const v4f*)(ArowL0 + k0 + 4);
      const v4f x2 = *(const v4f*)(ArowL0 + k1);
      const v4f x3 = *(const v4f*)(ArowL0 + k1 + 4);
      #pragma unroll
      for (int i = 0; i < 4; ++i) {
        a0[i]      = (_Float16)fmaxf(x0[i] + ba0[i], 0.f);
        a0[4 + i]  = (_Float16)fmaxf(x1[i] + ba1[i], 0.f);
        a0[8 + i]  = (_Float16)fmaxf(x2[i] + bb0[i], 0.f);
        a0[12 + i] = (_Float16)fmaxf(x3[i] + bb1[i], 0.f);
      }
    }
    {
      const v4f x0 = *(const v4f*)(ArowL1 + k0);
      const v4f x1 = *(const v4f*)(ArowL1 + k0 + 4);
      const v4f x2 = *(const v4f*)(ArowL1 + k1);
      const v4f x3 = *(const v4f*)(ArowL1 + k1 + 4);
      #pragma unroll
      for (int i = 0; i < 4; ++i) {
        a1[i]      = (_Float16)fmaxf(x0[i] + ba0[i], 0.f);
        a1[4 + i]  = (_Float16)fmaxf(x1[i] + ba1[i], 0.f);
        a1[8 + i]  = (_Float16)fmaxf(x2[i] + bb0[i], 0.f);
        a1[12 + i] = (_Float16)fmaxf(x3[i] + bb1[i], 0.f);
      }
    }

    #pragma unroll
    for (int ct = 0; ct < 8; ++ct) {
      const int col = chalf * 128 + ct * 16 + nl;
      const v8h bl = *(const v8h*)(W2t + (size_t)col * DQ + kb + koff);
      const v8h bh = *(const v8h*)(W2t + (size_t)col * DQ + kb + 16 + koff);
      const v16h bf = __builtin_shufflevector(bl, bh, 0, 1, 2, 3, 4, 5, 6, 7, 8, 9, 10, 11, 12, 13, 14, 15);
      acc[0][ct] = __builtin_amdgcn_wmma_f32_16x16x32_f16(
          false, a0, false, bf, (short)0, acc[0][ct], false, false);
      acc[1][ct] = __builtin_amdgcn_wmma_f32_16x16x32_f16(
          false, a1, false, bf, (short)0, acc[1][ct], false, false);
    }
  }

  // All waves must finish reading sA/sB before h2 overwrites them (union).
  __syncthreads();

  // ---- epilogue 1: relu(acc + b2) -> f16 into LDS (h2: ROWS x C1) ----
  #pragma unroll
  for (int rtl = 0; rtl < 2; ++rtl) {
    const int rt = rtb + rtl;
    #pragma unroll
    for (int ct = 0; ct < 8; ++ct) {
      const int col = chalf * 128 + ct * 16 + nl;
      #pragma unroll
      for (int j = 0; j < 8; ++j) {
        const int row = rt * 16 + j + 8 * lh;    // C-layout: VGPR j -> M = j (+8 hi-half)
        h2s[row * H2S + col] = (_Float16)fmaxf(acc[rtl][ct][j] + b2r[ct], 0.f);
      }
    }
  }
  __syncthreads();

  // ---- GEMM2: h2(128x256) @ W3t(256x16, 4 valid cols); wave w -> row-tile w ----
  v8f acc2;
  #pragma unroll
  for (int e = 0; e < 8; ++e) acc2[e] = 0.f;

  const _Float16* hrow = h2s + (size_t)(w * 16 + nl) * H2S;
  #pragma unroll
  for (int ks = 0; ks < C1 / 32; ++ks) {
    const int kb = ks * 32;
    const v8h c0 = *(const v8h*)(hrow + kb + koff);
    const v8h c1 = *(const v8h*)(hrow + kb + 16 + koff);
    v16h af;
    #pragma unroll
    for (int i = 0; i < 8; ++i) { af[i] = c0[i]; af[8 + i] = c1[i]; }
    const v8h bl = *(const v8h*)(W3t + nl * C1 + kb + koff);
    const v8h bh = *(const v8h*)(W3t + nl * C1 + kb + 16 + koff);
    const v16h bf = __builtin_shufflevector(bl, bh, 0, 1, 2, 3, 4, 5, 6, 7, 8, 9, 10, 11, 12, 13, 14, 15);
    acc2 = __builtin_amdgcn_wmma_f32_16x16x32_f16(
        false, af, false, bf, (short)0, acc2, false, false);
  }

  // ---- final store: out[b][n][m][o] with o = nl (<4), n = n0 + w ----
  if (nl < 4) {
    const float bias = b3[nl];
    #pragma unroll
    for (int j = 0; j < 8; ++j) {
      const int M = j + 8 * lh;                  // m_local
      const int n = n0 + w;
      const int m = m0 + M;
      out[(((size_t)(b * NQ + n)) * NQ + m) * 4 + nl] = acc2[j] + bias;
    }
  }
}

// ---------------------------------------------------------------------------
extern "C" void kernel_launch(void* const* d_in, const int* in_sizes, int n_in,
                              void* d_out, int out_size, void* d_ws, size_t ws_size,
                              hipStream_t stream) {
  const float* feat = (const float*)d_in[0];
  const float* pos  = (const float*)d_in[1];
  const float* W1   = (const float*)d_in[2];
  const float* b1   = (const float*)d_in[3];
  const float* W2   = (const float*)d_in[4];
  const float* b2   = (const float*)d_in[5];
  const float* W3   = (const float*)d_in[6];
  const float* b3   = (const float*)d_in[7];
  float* out = (float*)d_out;

  char* ws = (char*)d_ws;
  float*    Arows = (float*)ws;                                    // 2 MB
  float*    Brows = (float*)(ws + (size_t)2 * 1024 * 1024);        // 2 MB
  _Float16* W2t   = (_Float16*)(ws + (size_t)4 * 1024 * 1024);     // 256 KB
  _Float16* W3t   = (_Float16*)(ws + (size_t)4 * 1024 * 1024 + 256 * 1024); // 8 KB

  prep_rows<<<BQ * NQ, 256, 0, stream>>>(feat, pos, W1, b1, Arows, Brows);
  prep_w2t<<<(C1 * DQ) / 256, 256, 0, stream>>>(W2, W2t);
  prep_w3t<<<(16 * C1) / 256, 256, 0, stream>>>(W3, W3t);

  const int nblk = BQ * (NQ / TN) * (NQ / TM);        // 1024
  const size_t tiles_b = (size_t)(16 + 8) * TS * sizeof(float);        // 49,536
  const size_t h2_b    = (size_t)ROWS * H2S * sizeof(_Float16);        // 67,584
  const size_t ldsz    = tiles_b > h2_b ? tiles_b : h2_b;              // 67,584
  fused<<<nblk, 256, ldsz, stream>>>(Arows, Brows, W2t, W3t, b2, b3, out);
}
